// AttentionLayer_11596411699556
// MI455X (gfx1250) — matrix operand order repaired and checked by reading, not compiled, not run on hardware
//
#include <hip/hip_runtime.h>
#include <math.h>

typedef __bf16 bf16_t;
typedef __attribute__((ext_vector_type(8)))  __bf16 v8bf;
typedef __attribute__((ext_vector_type(16))) __bf16 v16bf;
typedef __attribute__((ext_vector_type(8)))  float   v8f;

#define SEQ   2048
#define EMB   1024
#define NH    16
#define HD    64
#define KVD   256    // EMB / N_KV_GROUPS
#define NCAT  1536   // q(1024) + k(256) + v(256)

union ABu { v16bf v; v8bf h[2]; };

__device__ __forceinline__ v8f wmma_bf16(v16bf a, v16bf b, v8f c) {
  // D = A(16x32 bf16) * B(32x16 bf16) + C(16x16 f32)
  return __builtin_amdgcn_wmma_f32_16x16x32_bf16(false, a, false, b, (short)0, c, false, false);
}

// ---------------- conversion / weight prep ----------------
__global__ void k_cvt(const float* __restrict__ src, bf16_t* __restrict__ dst, int n) {
  int i = blockIdx.x * blockDim.x + threadIdx.x;
  if (i < n) dst[i] = (bf16_t)src[i];
}

// W_qkv_t[n][k] = Wq|Wk|Wv [k][n']  (n over 1536 concatenated output cols)
__global__ void k_wcat(const float* __restrict__ Wq, const float* __restrict__ Wk,
                       const float* __restrict__ Wv, bf16_t* __restrict__ Wt) {
  int i = blockIdx.x * blockDim.x + threadIdx.x;
  if (i >= EMB * NCAT) return;
  int k = i / NCAT, n = i % NCAT;   // reads coalesced along n
  float v;
  if (n < EMB)            v = Wq[k * EMB + n];
  else if (n < EMB + KVD) v = Wk[k * KVD + (n - EMB)];
  else                    v = Wv[k * KVD + (n - EMB - KVD)];
  Wt[n * EMB + k] = (bf16_t)v;
}

__global__ void k_wot(const float* __restrict__ Wo, bf16_t* __restrict__ Wt) {
  int i = blockIdx.x * blockDim.x + threadIdx.x;
  if (i >= EMB * EMB) return;
  int k = i / EMB, n = i % EMB;
  Wt[n * EMB + k] = (bf16_t)Wo[k * EMB + n];
}

// ---------------- fused QKV GEMM + RoPE ----------------
// One wave computes a 32(M) x 64(N) tile of hs @ [Wq|Wk|Wv]:
// two 16-row A fragments share each streaming B fragment (1.5 loads/WMMA).
// Load-and-consume body: the compiler's unroll+clause scheduling pipelines this
// without doubling live fragment state (explicit double-buffering spills).
__global__ void k_gemm_qkv(const bf16_t* __restrict__ A, const bf16_t* __restrict__ Bt,
                           bf16_t* __restrict__ Q, bf16_t* __restrict__ K,
                           bf16_t* __restrict__ Vt) {
  const int lane = threadIdx.x & 31;
  const int wave = threadIdx.x >> 5;
  const int gw   = blockIdx.x * 8 + wave;
  const int mt = gw / (NCAT / 64);       // 64 row tiles of 32
  const int nt64 = gw % (NCAT / 64);     // 24 col tiles of 64
  const int m0 = mt * 32, n0 = nt64 * 64;
  const int h = lane >> 4, n16 = lane & 15;

  v8f acc[2][4] = {};
  const bf16_t* ar0 = A + (size_t)(m0 + n16) * EMB;        // rows m0..m0+15
  const bf16_t* ar1 = A + (size_t)(m0 + 16 + n16) * EMB;   // rows m0+16..m0+31
  for (int k = 0; k < EMB; k += 32) {
    ABu a0, a1;
    a0.h[0] = *(const v8bf*)(ar0 + k + h * 16);
    a0.h[1] = *(const v8bf*)(ar0 + k + 8 + h * 16);
    a1.h[0] = *(const v8bf*)(ar1 + k + h * 16);
    a1.h[1] = *(const v8bf*)(ar1 + k + 8 + h * 16);
#pragma unroll
    for (int nt = 0; nt < 4; ++nt) {
      v16bf b = *(const v16bf*)(Bt + (size_t)(n0 + nt * 16 + n16) * EMB + k + h * 16);
      acc[0][nt] = wmma_bf16(a0.v, b, acc[0][nt]);
      acc[1][nt] = wmma_bf16(a1.v, b, acc[1][nt]);
    }
  }

#pragma unroll
  for (int ms = 0; ms < 2; ++ms) {
    const int mb = m0 + ms * 16;
    if (n0 < EMB) {                     // ---- Q: RoPE then store bf16 row-major
#pragma unroll
      for (int nt = 0; nt < 4; ++nt) {
        int c = n0 + nt * 16 + n16;
        int d = c & 63;
        int idx = (d < 32) ? d : d - 32;
        float inv = __powf(10000.f, -(float)(2 * idx) * (1.f / 64.f));
#pragma unroll
        for (int e = 0; e < 8; ++e) {
          int t = mb + e + 8 * h;       // D layout: row = e + 8*(lane>>4)
          float sn, cs; __sincosf((float)t * inv, &sn, &cs);
          float x = acc[ms][nt][e], xp = acc[ms][nt ^ 2][e];
          float r = (d < 32) ? (x * cs - xp * sn) : (x * cs + xp * sn);
          Q[(size_t)t * EMB + c] = (bf16_t)r;
        }
      }
    } else if (n0 < EMB + KVD) {        // ---- K: RoPE then store bf16 row-major
#pragma unroll
      for (int nt = 0; nt < 4; ++nt) {
        int ck = n0 + nt * 16 + n16 - EMB;
        int d = ck & 63;
        int idx = (d < 32) ? d : d - 32;
        float inv = __powf(10000.f, -(float)(2 * idx) * (1.f / 64.f));
#pragma unroll
        for (int e = 0; e < 8; ++e) {
          int t = mb + e + 8 * h;
          float sn, cs; __sincosf((float)t * inv, &sn, &cs);
          float x = acc[ms][nt][e], xp = acc[ms][nt ^ 2][e];
          float r = (d < 32) ? (x * cs - xp * sn) : (x * cs + xp * sn);
          K[(size_t)t * KVD + ck] = (bf16_t)r;
        }
      }
    } else {                            // ---- V: store TRANSPOSED Vt[d][t]
#pragma unroll
      for (int nt = 0; nt < 4; ++nt) {
        int cv = n0 + nt * 16 + n16 - (EMB + KVD);
#pragma unroll
        for (int e = 0; e < 8; ++e) {
          int t = mb + e + 8 * h;
          Vt[(size_t)cv * SEQ + t] = (bf16_t)acc[ms][nt][e];
        }
      }
    }
  }
}

// ---------------- causal flash attention ----------------
// One wave per (head, 16-row q tile). 32-key blocks, online softmax.
__global__ void k_flash(const bf16_t* __restrict__ Q, const bf16_t* __restrict__ K,
                        const bf16_t* __restrict__ Vt, bf16_t* __restrict__ O) {
  __shared__ alignas(32) bf16_t Pl[8][16 * 32];   // per-wave P tile (16 q x 32 k)
  const int lane = threadIdx.x & 31;
  const int wave = threadIdx.x >> 5;
  const int gw = blockIdx.x * 8 + wave;
  const int hidx = gw >> 7;            // head 0..15
  const int mt = gw & 127;
  const int m0 = mt * 16;
  const int kvh = hidx >> 2;           // GQA: kv head = head / 4
  const int h = lane >> 4, n16 = lane & 15;
  const float scale = 0.125f;          // 1/sqrt(64)

  float m_r[8], l_r[8], alpha[8];
  v8f o_acc[4] = {};
#pragma unroll
  for (int r = 0; r < 8; ++r) { m_r[r] = -3.0e38f; l_r[r] = 0.f; }

  // Q fragments (16x64 -> two 16x32 A-frags), rows = lane%16
  ABu q0, q1;
  const bf16_t* qp = Q + (size_t)(m0 + n16) * EMB + hidx * HD;
  q0.h[0] = *(const v8bf*)(qp + h * 16);
  q0.h[1] = *(const v8bf*)(qp + 8 + h * 16);
  q1.h[0] = *(const v8bf*)(qp + 32 + h * 16);
  q1.h[1] = *(const v8bf*)(qp + 40 + h * 16);

  bf16_t* pl = &Pl[wave][0];
  const int nblocks = (m0 + 16 + 31) >> 5;
  for (int kb = 0; kb < nblocks; ++kb) {
    const int ks = kb * 32;
    // prefetch next key block's K/V streams (global_prefetch_b8)
    if (kb + 1 < nblocks) {
      __builtin_prefetch(K + (size_t)(ks + 32 + n16) * KVD + kvh * HD, 0, 0);
      __builtin_prefetch(Vt + (size_t)(kvh * HD + n16) * SEQ + ks + 32, 0, 0);
    }
    // ---- S = Q @ K^T for two 16-key subtiles
    v8f s[2];
#pragma unroll
    for (int j = 0; j < 2; ++j) {
      const bf16_t* kp = K + (size_t)(ks + j * 16 + n16) * KVD + kvh * HD;
      v16bf b0 = *(const v16bf*)(kp + h * 16);        // k-dims 0..31
      v16bf b1 = *(const v16bf*)(kp + 32 + h * 16);   // k-dims 32..63
      v8f sj = {};
      sj = wmma_bf16(q0.v, b0, sj);
      sj = wmma_bf16(q1.v, b1, sj);
      s[j] = sj;
    }
    // ---- scale + causal mask (element (r): q row = m0+r+8h, key = ks+j*16+n16)
#pragma unroll
    for (int j = 0; j < 2; ++j)
#pragma unroll
      for (int r = 0; r < 8; ++r) {
        int trow = m0 + r + 8 * h;
        int key = ks + j * 16 + n16;
        float v = s[j][r] * scale;
        s[j][r] = (key <= trow) ? v : -3.0e38f;
      }
    // ---- online softmax (row reductions stay inside 16-lane halves)
#pragma unroll
    for (int r = 0; r < 8; ++r) {
      float v = fmaxf(s[0][r], s[1][r]);
#pragma unroll
      for (int off = 8; off; off >>= 1) v = fmaxf(v, __shfl_xor(v, off, 32));
      float mn = fmaxf(m_r[r], v);
      alpha[r] = __expf(m_r[r] - mn);
      m_r[r] = mn;
      float p0 = __expf(s[0][r] - mn);
      float p1 = __expf(s[1][r] - mn);
      s[0][r] = p0; s[1][r] = p1;
      float rs = p0 + p1;
#pragma unroll
      for (int off = 8; off; off >>= 1) rs += __shfl_xor(rs, off, 32);
      l_r[r] = l_r[r] * alpha[r] + rs;
    }
#pragma unroll
    for (int nt = 0; nt < 4; ++nt)
#pragma unroll
      for (int r = 0; r < 8; ++r) o_acc[nt][r] *= alpha[r];
    // ---- transpose P through LDS into A-fragment layout
#pragma unroll
    for (int j = 0; j < 2; ++j)
#pragma unroll
      for (int r = 0; r < 8; ++r)
        pl[(r + 8 * h) * 32 + j * 16 + n16] = (bf16_t)s[j][r];
    asm volatile("s_wait_dscnt 0x0" ::: "memory");   // cross-lane LDS dependency
    ABu pa;
    pa.h[0] = *(const v8bf*)(pl + n16 * 32 + h * 16);
    pa.h[1] = *(const v8bf*)(pl + n16 * 32 + 8 + h * 16);
    // ---- O += P @ V  (Vt is d-major so B-frag is contiguous)
#pragma unroll
    for (int nt = 0; nt < 4; ++nt) {
      const bf16_t* vp = Vt + (size_t)(kvh * HD + nt * 16 + n16) * SEQ + ks;
      v16bf vb = *(const v16bf*)(vp + h * 16);
      o_acc[nt] = wmma_bf16(pa.v, vb, o_acc[nt]);
    }
  }
  // ---- normalize and store O (bf16, row-major [seq][emb])
#pragma unroll
  for (int nt = 0; nt < 4; ++nt)
#pragma unroll
    for (int r = 0; r < 8; ++r) {
      int t = m0 + r + 8 * h;
      O[(size_t)t * EMB + hidx * HD + nt * 16 + n16] = (bf16_t)(o_acc[nt][r] / l_r[r]);
    }
}

// ---------------- output projection: out = O @ Wo (f32 store) ----------------
// Same 32x64-tile structure as the QKV GEMM.
__global__ void k_gemm_out(const bf16_t* __restrict__ A, const bf16_t* __restrict__ Bt,
                           float* __restrict__ out) {
  const int lane = threadIdx.x & 31;
  const int wave = threadIdx.x >> 5;
  const int gw = blockIdx.x * 8 + wave;
  const int mt = gw / (EMB / 64);        // 64 row tiles of 32
  const int nt64 = gw % (EMB / 64);      // 16 col tiles of 64
  const int m0 = mt * 32, n0 = nt64 * 64;
  const int h = lane >> 4, n16 = lane & 15;

  v8f acc[2][4] = {};
  const bf16_t* ar0 = A + (size_t)(m0 + n16) * EMB;
  const bf16_t* ar1 = A + (size_t)(m0 + 16 + n16) * EMB;
  for (int k = 0; k < EMB; k += 32) {
    ABu a0, a1;
    a0.h[0] = *(const v8bf*)(ar0 + k + h * 16);
    a0.h[1] = *(const v8bf*)(ar0 + k + 8 + h * 16);
    a1.h[0] = *(const v8bf*)(ar1 + k + h * 16);
    a1.h[1] = *(const v8bf*)(ar1 + k + 8 + h * 16);
#pragma unroll
    for (int nt = 0; nt < 4; ++nt) {
      v16bf b = *(const v16bf*)(Bt + (size_t)(n0 + nt * 16 + n16) * EMB + k + h * 16);
      acc[0][nt] = wmma_bf16(a0.v, b, acc[0][nt]);
      acc[1][nt] = wmma_bf16(a1.v, b, acc[1][nt]);
    }
  }
#pragma unroll
  for (int ms = 0; ms < 2; ++ms)
#pragma unroll
    for (int nt = 0; nt < 4; ++nt)
#pragma unroll
      for (int e = 0; e < 8; ++e) {
        int t = m0 + ms * 16 + e + 8 * h;
        out[(size_t)t * EMB + n0 + nt * 16 + n16] = acc[ms][nt][e];
      }
}

extern "C" void kernel_launch(void* const* d_in, const int* in_sizes, int n_in,
                              void* d_out, int out_size, void* d_ws, size_t ws_size,
                              hipStream_t stream) {
  (void)in_sizes; (void)n_in; (void)out_size; (void)ws_size;
  const float* hs = (const float*)d_in[0];
  const float* Wq = (const float*)d_in[1];
  const float* Wk = (const float*)d_in[2];
  const float* Wv = (const float*)d_in[3];
  const float* Wo = (const float*)d_in[4];
  float* out = (float*)d_out;

  char* ws = (char*)d_ws;
  size_t off = 0;
  auto alloc = [&](size_t bytes) { void* p = ws + off; off += (bytes + 255) & ~(size_t)255; return p; };
  bf16_t* hsb  = (bf16_t*)alloc((size_t)SEQ * EMB * 2);
  bf16_t* Wcat = (bf16_t*)alloc((size_t)NCAT * EMB * 2);
  bf16_t* Wot  = (bf16_t*)alloc((size_t)EMB * EMB * 2);
  bf16_t* Qb   = (bf16_t*)alloc((size_t)SEQ * EMB * 2);
  bf16_t* Kb   = (bf16_t*)alloc((size_t)SEQ * KVD * 2);
  bf16_t* Vtb  = (bf16_t*)alloc((size_t)KVD * SEQ * 2);
  bf16_t* Ob   = (bf16_t*)alloc((size_t)SEQ * EMB * 2);

  k_cvt<<<(SEQ * EMB + 255) / 256, 256, 0, stream>>>(hs, hsb, SEQ * EMB);
  k_wcat<<<(EMB * NCAT + 255) / 256, 256, 0, stream>>>(Wq, Wk, Wv, Wcat);
  k_wot<<<(EMB * EMB + 255) / 256, 256, 0, stream>>>(Wo, Wot);
  // 64 row-tiles x 24 col-tiles = 1536 waves, 8 waves/block
  k_gemm_qkv<<<192, 256, 0, stream>>>(hsb, Wcat, Qb, Kb, Vtb);
  // 16 heads x 128 q-tiles = 2048 waves
  k_flash<<<256, 256, 0, stream>>>(Qb, Kb, Vtb, Ob);
  // 64 x 16 = 1024 waves
  k_gemm_out<<<128, 256, 0, stream>>>(Ob, Wot, out);
}
